// rnn_models_77764677862241
// MI455X (gfx1250) — matrix operand order repaired and checked by reading, not compiled, not run on hardware
//
#include <hip/hip_runtime.h>
#include <hip/hip_bf16.h>
#include <cstddef>

// ---------------- problem constants (match reference) ----------------
#define N_UNITS 512
#define INPUT_DIM 128
#define OUT_CLASSES 10
#define BATCH 128
#define T_STEPS 1024
#define EPS 0.01f
#define BETA 0.8f
#define GAMMA 0.01f
#define ALPHA 1.0f

// ---------------- vector types for WMMA ----------------
typedef __bf16 bf16x16 __attribute__((ext_vector_type(16)));
typedef __bf16 bf16x8  __attribute__((ext_vector_type(8)));
typedef __bf16 bf16x4  __attribute__((ext_vector_type(4)));
typedef float  f32x8   __attribute__((ext_vector_type(8)));

__device__ __forceinline__ f32x8 wmma_bf16(bf16x16 a, bf16x16 b, f32x8 c) {
  // D = A(16x32) * B(32x16) + C, f32 accumulate
  return __builtin_amdgcn_wmma_f32_16x16x32_bf16(
      /*neg_a=*/false, a, /*neg_b=*/false, b,
      /*c_mod=*/(short)0, c, /*reuse_a=*/false, /*reuse_b=*/false);
}

// A-operand (16xK tile, bf16) from LDS, row-major with padded stride.
// Lane layout: lanes 0-15 -> M=lane, K = k0..k0+15 contiguous
//              lanes 16-31 -> M=lane-16, K = k0+16..k0+31.
// kb must already include the +16 for the upper half-wave.
__device__ __forceinline__ bf16x16 load_a_lds(const __bf16* buf, int stride,
                                              int m, int kb) {
  const __bf16* p = buf + m * stride + kb;
  bf16x8 lo = *(const bf16x8*)(p);
  bf16x8 hi = *(const bf16x8*)(p + 8);
  return __builtin_shufflevector(lo, hi, 0, 1, 2, 3, 4, 5, 6, 7,
                                 8, 9, 10, 11, 12, 13, 14, 15);
}

// B-operand (Kx16 tile, bf16) from global. Matrix stored K-contiguous per
// column: Mt[n * ldk + k] == Mat[k][n].  Lane layout: N = n0 + lane%16,
// K = k0 + 16*(lane>=16) + j, j = 0..15 contiguous -> two b128 loads.
// Base = Mt + off (uniform SGPR, `off` is an opaque runtime-zero offset that
// defeats hoist+spill); idx = 32-bit divergent voffset -> GVS saddr form.
__device__ __forceinline__ bf16x16 load_b_glb(const __bf16* Mt, size_t off,
                                              int ldk, int n0, int k0,
                                              int lmod, int lhalf) {
  const __bf16* base = Mt + off;  // uniform scalar base (one s_add per step)
  const unsigned idx =
      (unsigned)((n0 + lmod) * ldk + k0 + lhalf * 16);  // loop-invariant voffset
  bf16x8 lo = *(const bf16x8*)(base + idx);
  bf16x8 hi = *(const bf16x8*)(base + idx + 8);
  return __builtin_shufflevector(lo, hi, 0, 1, 2, 3, 4, 5, 6, 7,
                                 8, 9, 10, 11, 12, 13, 14, 15);
}

// ---------------- kernel 1: build A^T, W^T, E_w^T in bf16 ----------------
// At[n*512+k] = A[k][n] with A = b*(B-B^T)+(1-b)*(B+B^T)-g*I  (same for W/C)
// Ewt[n*128+i] = E_w[i][n]
__global__ __launch_bounds__(256) void prep_kernel(
    const float* __restrict__ B, const float* __restrict__ C,
    const float* __restrict__ Ew, __bf16* __restrict__ At,
    __bf16* __restrict__ Wt, __bf16* __restrict__ Ewt) {
  int i = blockIdx.x * 256 + threadIdx.x;
  if (i < N_UNITS * N_UNITS) {
    int n = i >> 9;       // row of transposed store
    int k = i & 511;      // contiguous K
    float bkn = B[k * N_UNITS + n], bnk = B[n * N_UNITS + k];
    float ckn = C[k * N_UNITS + n], cnk = C[n * N_UNITS + k];
    float diag = (k == n) ? GAMMA : 0.0f;
    At[i] = (__bf16)(BETA * (bkn - bnk) + (1.0f - BETA) * (bkn + bnk) - diag);
    Wt[i] = (__bf16)(BETA * (ckn - cnk) + (1.0f - BETA) * (ckn + cnk) - diag);
  } else {
    int j = i - N_UNITS * N_UNITS;
    if (j < N_UNITS * INPUT_DIM) {
      int n = j >> 7;
      int k = j & 127;
      Ewt[j] = (__bf16)Ew[k * N_UNITS + n];
    }
  }
}

// ---------------- kernel 2: per-batch-tile recurrent scan ----------------
// 8 workgroups x 512 threads (16 waves). Each workgroup owns 16 batch rows.
// Each wave owns two 16x16 N-tiles of h, kept in f32 WMMA accumulators.
// bf16 shadow of h lives in LDS (padded) for next step's A-operand.
// A/W/Ew B-operands stream from L2 every step via global_load_b128 (saddr).
#define HB_STRIDE (N_UNITS + 8)    // 520 bf16 -> 1040B row (16B aligned, bank-skewed)
#define XB_STRIDE (INPUT_DIM + 8)  // 136 bf16 -> 272B row

__global__ __launch_bounds__(512, 1) void scan_kernel(
    const float* __restrict__ x, const __bf16* __restrict__ At,
    const __bf16* __restrict__ Wt, const __bf16* __restrict__ Ewt,
    const float* __restrict__ Eb, float* __restrict__ hout) {
  __shared__ alignas(16) __bf16 hb[16 * HB_STRIDE];  // bf16 h, A-operand source
  __shared__ alignas(16) __bf16 xb[16 * XB_STRIDE];  // bf16 x_t tile

  const int tid = threadIdx.x;
  const int lane = tid & 31;
  const int wave = tid >> 5;          // 0..15
  const int lmod = lane & 15;
  const int lhalf = (lane >> 4) & 1;  // 0: lanes 0-15, 1: lanes 16-31
  const int b0 = blockIdx.x * 16;     // batch rows of this workgroup
  const int n0a = wave * 32;          // first N tile
  const int n0b = n0a + 16;           // second N tile

  // zero the bf16 h shadow (h0 = 0)
  for (int i = tid; i < 16 * HB_STRIDE; i += 512) hb[i] = (__bf16)0.0f;

  const float eb_a = Eb[n0a + lmod];
  const float eb_b = Eb[n0b + lmod];

  f32x8 h0 = {}, h1 = {};  // persistent f32 state (C/D layout)

  // x staging coordinates: each thread moves 4 floats of the 16x128 slice
  const int xrow = tid >> 5;
  const int xcol = (tid & 31) * 4;
  const size_t xrow_base = ((size_t)(b0 + xrow)) * T_STEPS * INPUT_DIM + xcol;

  // Opaque element offset (runtime value 0, but the optimizer cannot know
  // that): makes the weight-stream loads loop-variant so they are re-issued
  // from L2 each step instead of being hoisted and spilled to scratch.
  size_t off = 0;

  for (int t = 0; t < T_STEPS; ++t) {
    // ---- stage x[:, t, :] tile into LDS as bf16 ----
    {
      const float4 xv = *(const float4*)&x[xrow_base + (size_t)t * INPUT_DIM];
      bf16x4 xc;
      xc[0] = (__bf16)xv.x; xc[1] = (__bf16)xv.y;
      xc[2] = (__bf16)xv.z; xc[3] = (__bf16)xv.w;
      *(bf16x4*)&xb[xrow * XB_STRIDE + xcol] = xc;
      if (t + 1 < T_STEPS)  // gfx1250 global_prefetch_b8 for next timestep
        __builtin_prefetch(&x[xrow_base + (size_t)(t + 1) * INPUT_DIM], 0, 1);
    }
    __syncthreads();  // xb ready; hb (from previous step) ready

    f32x8 accW0 = {}, accW1 = {}, accA0 = {}, accA1 = {};

    // ---- z contribution: x_t (16x128) @ E_w (128x512) ----
#pragma unroll
    for (int k0 = 0; k0 < INPUT_DIM; k0 += 32) {
      bf16x16 a = load_a_lds(xb, XB_STRIDE, lmod, k0 + lhalf * 16);
      accW0 = wmma_bf16(a, load_b_glb(Ewt, off, INPUT_DIM, n0a, k0, lmod, lhalf), accW0);
      accW1 = wmma_bf16(a, load_b_glb(Ewt, off, INPUT_DIM, n0b, k0, lmod, lhalf), accW1);
    }
    // ---- recurrent terms: h (16x512) @ W and h @ A (L2-resident B-operands) ----
#pragma unroll
    for (int k0 = 0; k0 < N_UNITS; k0 += 32) {
      bf16x16 a = load_a_lds(hb, HB_STRIDE, lmod, k0 + lhalf * 16);
      accW0 = wmma_bf16(a, load_b_glb(Wt, off, N_UNITS, n0a, k0, lmod, lhalf), accW0);
      accW1 = wmma_bf16(a, load_b_glb(Wt, off, N_UNITS, n0b, k0, lmod, lhalf), accW1);
      accA0 = wmma_bf16(a, load_b_glb(At, off, N_UNITS, n0a, k0, lmod, lhalf), accA0);
      accA1 = wmma_bf16(a, load_b_glb(At, off, N_UNITS, n0b, k0, lmod, lhalf), accA1);
    }

    // ---- h := h + eps*alpha*(h@A) + eps*tanh(h@W + z + E_b) ----
#pragma unroll
    for (int r = 0; r < 8; ++r) {
      h0[r] = h0[r] + (EPS * ALPHA) * accA0[r] + EPS * tanhf(accW0[r] + eb_a);
      h1[r] = h1[r] + (EPS * ALPHA) * accA1[r] + EPS * tanhf(accW1[r] + eb_b);
    }

    __syncthreads();  // everyone finished reading hb of this step
    // write bf16 shadow for next step; C/D layout -> row-major LDS
#pragma unroll
    for (int r = 0; r < 8; ++r) {
      const int m = r + 8 * lhalf;
      hb[m * HB_STRIDE + n0a + lmod] = (__bf16)h0[r];
      hb[m * HB_STRIDE + n0b + lmod] = (__bf16)h1[r];
    }

    // Defeat loop-invariant hoisting of the B-operand streams (no-op at runtime).
    asm volatile("" : "+s"(off));
  }

  // ---- spill final f32 h state to workspace ----
#pragma unroll
  for (int r = 0; r < 8; ++r) {
    const int m = r + 8 * lhalf;
    hout[(size_t)(b0 + m) * N_UNITS + n0a + lmod] = h0[r];
    hout[(size_t)(b0 + m) * N_UNITS + n0b + lmod] = h1[r];
  }
}

// ---------------- kernel 3: tiny readout h_final @ D_w + D_b ----------------
__global__ __launch_bounds__(256) void out_kernel(
    const float* __restrict__ hf, const float* __restrict__ Dw,
    const float* __restrict__ Db, float* __restrict__ out) {
  int i = blockIdx.x * 256 + threadIdx.x;
  if (i >= BATCH * OUT_CLASSES) return;
  int b = i / OUT_CLASSES, o = i % OUT_CLASSES;
  float acc = Db[o];
#pragma unroll 8
  for (int k = 0; k < N_UNITS; ++k)
    acc += hf[(size_t)b * N_UNITS + k] * Dw[k * OUT_CLASSES + o];
  out[i] = acc;
}

// ---------------- host entry ----------------
extern "C" void kernel_launch(void* const* d_in, const int* in_sizes, int n_in,
                              void* d_out, int out_size, void* d_ws,
                              size_t ws_size, hipStream_t stream) {
  (void)in_sizes; (void)n_in; (void)out_size; (void)ws_size;
  const float* x  = (const float*)d_in[0];  // [128,1024,128]
  const float* B  = (const float*)d_in[1];  // [512,512]
  const float* C  = (const float*)d_in[2];  // [512,512]
  const float* Ew = (const float*)d_in[3];  // [128,512]
  const float* Eb = (const float*)d_in[4];  // [512]
  const float* Dw = (const float*)d_in[5];  // [512,10]
  const float* Db = (const float*)d_in[6];  // [10]
  float* out = (float*)d_out;               // [128,10]

  // workspace carve (~1.44 MB)
  char* ws = (char*)d_ws;
  __bf16* At  = (__bf16*)ws;  ws += (size_t)N_UNITS * N_UNITS * sizeof(__bf16);
  __bf16* Wt  = (__bf16*)ws;  ws += (size_t)N_UNITS * N_UNITS * sizeof(__bf16);
  __bf16* Ewt = (__bf16*)ws;  ws += (size_t)N_UNITS * INPUT_DIM * sizeof(__bf16);
  float*  hf  = (float*)ws;   ws += (size_t)BATCH * N_UNITS * sizeof(float);

  {  // prep: (512*512 for A&W) + (512*128 for Ewt) threads
    int total = N_UNITS * N_UNITS + N_UNITS * INPUT_DIM;
    prep_kernel<<<(total + 255) / 256, 256, 0, stream>>>(B, C, Ew, At, Wt, Ewt);
  }
  scan_kernel<<<BATCH / 16, 512, 0, stream>>>(x, At, Wt, Ewt, Eb, hf);
  out_kernel<<<(BATCH * OUT_CLASSES + 255) / 256, 256, 0, stream>>>(hf, Dw, Db, out);
}
